// TransformerDecoderLayer_17102559772672
// MI455X (gfx1250) — matrix operand order repaired and checked by reading, not compiled, not run on hardware
//
#include <hip/hip_runtime.h>

// ---------------------------------------------------------------------------
// Problem constants (from the reference)
// ---------------------------------------------------------------------------
#define D_   256
#define H_   8
#define HD_  32
#define NQ_  64
#define B_   64
#define TQ_  4096
#define TK_  49152
#define L_   64
#define FF_  1024

typedef __attribute__((ext_vector_type(16))) __bf16 v16bf;
typedef __attribute__((ext_vector_type(8)))  float  v8f;

__device__ __forceinline__ unsigned short f2bf(float f) {
  unsigned int u = __float_as_uint(f);
  u += 0x7fffu + ((u >> 16) & 1u);          // round-to-nearest-even
  return (unsigned short)(u >> 16);
}
__device__ __forceinline__ float bf2f(unsigned short h) {
  return __uint_as_float(((unsigned int)h) << 16);
}
// packed f32x2 -> bf16x2 (single v_cvt_pk_bf16_f32 when the builtin exists)
__device__ __forceinline__ unsigned int pkbf(float lo, float hi) {
#if __has_builtin(__builtin_amdgcn_cvt_pk_bf16_f32)
  auto r = __builtin_amdgcn_cvt_pk_bf16_f32(lo, hi);
  unsigned int u;
  __builtin_memcpy(&u, &r, sizeof(u));
  return u;
#else
  return (unsigned)f2bf(lo) | ((unsigned)f2bf(hi) << 16);
#endif
}
__device__ __forceinline__ float bfdot2(unsigned int a, unsigned int b) {
  float alo = __uint_as_float(a << 16);
  float ahi = __uint_as_float(a & 0xffff0000u);
  float blo = __uint_as_float(b << 16);
  float bhi = __uint_as_float(b & 0xffff0000u);
  return alo * blo + ahi * bhi;
}

// ---------------------------------------------------------------------------
// Weight convert: Wt[n*K + k] = bf16(W[k*N + n])   (transpose + downconvert)
// ---------------------------------------------------------------------------
__global__ __launch_bounds__(256) void wconv_kernel(
    const float* __restrict__ W, unsigned short* __restrict__ Wt, int K, int N) {
  const int i = blockIdx.x * 256 + threadIdx.x;
  if (i >= N * K) return;
  const int n = i / K;
  const int k = i - n * K;
  Wt[i] = f2bf(W[(size_t)k * N + n]);
}

// exclusive prefix sum of key_batch_cnt (B_=64 entries)
__global__ void prefix_kernel(const int* __restrict__ cnt, int* __restrict__ off) {
  const int t = threadIdx.x;
  int s = 0;
  for (int i = 0; i < t; ++i) s += cnt[i];
  off[t] = s;
}

// ---------------------------------------------------------------------------
// Generic bf16 WMMA GEMM:  out = epilogue( A1*W1^T [+ A2*W2^T] + bias )
//   Block tile: 128(M) x 128(N), K-step 32, 256 threads = 8 waves.
//   Wave grid 4x2; each wave owns 2 row-tiles x 4 col-tiles (8 accum tiles).
//   Software pipelined: global loads for step s+1 issued while step s computes.
//   FLAGS is a compile-time template parameter -> straight-line inner loop.
// ---------------------------------------------------------------------------
enum {
  F_ABF16   = 1,    // A matrices stored bf16 (else fp32, converted on stage-in)
  F_DUAL    = 2,    // accumulate A1*W1 + A2*W2 (same K)
  F_PERMA1  = 4,    // logical row r reads phys row (r%64)*64 + r/64  (bn<->nb)
  F_PERMA2  = 8,
  F_RELU    = 16,
  F_OBF16   = 32,   // bf16 output
  F_Q2      = 64,   // per-head concat column map: oc=(c/32)*64+colsel*32+c%32
  F_ADDPREV = 128,  // add bf16 value already stored at [r, oc+32] (k2 fuse)
  F_MASK    = 256   // multiply by rowmask[r]
};

#define BM 128
#define BN 128
#define BK 32

template <int FLAGS>
__global__ __launch_bounds__(256) void gemm_wmma_bf16(
    const void* __restrict__ A1, const void* __restrict__ A2,
    const unsigned short* __restrict__ Wt1, const unsigned short* __restrict__ Wt2,
    const float* __restrict__ bias1, const float* __restrict__ bias2,
    const unsigned char* __restrict__ rowmask,
    void* __restrict__ outp,
    int K, int colsel, int ldc)
{
  __shared__ __align__(16) unsigned short As[BM * BK];
  __shared__ __align__(16) unsigned short Bs[BN * BK];

  const int tid  = threadIdx.x;
  const int lane = tid & 31;
  const int wid  = tid >> 5;
  const int wr   = wid & 3;      // 2 row tiles: rows wr*32 + {0,16}
  const int wc   = wid >> 2;     // col half: cols wc*64 + 4 tiles of 16
  const int rowBase = blockIdx.y * BM;
  const int colBase = blockIdx.x * BN;

  const int steps1 = K / BK;
  const int nsteps = (FLAGS & F_DUAL) ? 2 * steps1 : steps1;

  // staging mapping: each thread moves 16 elements of A and 16 of B per step
  const int ar = tid >> 1;            // 0..127
  const int ac = (tid & 1) * 16;      // 0 / 16

  v8f acc[2][4];
#pragma unroll
  for (int rt = 0; rt < 2; ++rt)
#pragma unroll
    for (int j = 0; j < 4; ++j)
#pragma unroll
      for (int e = 0; e < 8; ++e) acc[rt][j][e] = 0.0f;

  uint4 ra0, ra1, rb0, rb1;   // pipeline registers (next tile)

  auto stageLoad = [&](int s) {
    const bool second = (FLAGS & F_DUAL) && (s >= steps1);
    const int kk = (second ? (s - steps1) : s) * BK;
    const void* A = second ? A2 : A1;
    const unsigned short* Wt = second ? Wt2 : Wt1;
    const bool perm = (FLAGS & (second ? F_PERMA2 : F_PERMA1)) != 0;
    int gr = rowBase + ar;
    if (perm) gr = (gr & 63) * 64 + (gr >> 6);
    if (FLAGS & F_ABF16) {
      const unsigned short* p = (const unsigned short*)A + (size_t)gr * K + kk + ac;
      ra0 = *(const uint4*)p;
      ra1 = *(const uint4*)(p + 8);
    } else {
      const float* p = (const float*)A + (size_t)gr * K + kk + ac;
      const float4 v0 = *(const float4*)p;
      const float4 v1 = *(const float4*)(p + 4);
      const float4 v2 = *(const float4*)(p + 8);
      const float4 v3 = *(const float4*)(p + 12);
      ra0.x = pkbf(v0.x, v0.y);
      ra0.y = pkbf(v0.z, v0.w);
      ra0.z = pkbf(v1.x, v1.y);
      ra0.w = pkbf(v1.z, v1.w);
      ra1.x = pkbf(v2.x, v2.y);
      ra1.y = pkbf(v2.z, v2.w);
      ra1.z = pkbf(v3.x, v3.y);
      ra1.w = pkbf(v3.z, v3.w);
    }
    const unsigned short* q = Wt + (size_t)(colBase + ar) * K + kk + ac;
    rb0 = *(const uint4*)q;
    rb1 = *(const uint4*)(q + 8);
  };

  stageLoad(0);

  for (int s = 0; s < nsteps; ++s) {
    // commit pipeline regs to LDS
    *(uint4*)&As[ar * BK + ac]     = ra0;
    *(uint4*)&As[ar * BK + ac + 8] = ra1;
    *(uint4*)&Bs[ar * BK + ac]     = rb0;
    *(uint4*)&Bs[ar * BK + ac + 8] = rb1;
    __syncthreads();

    // issue next tile's global loads while this tile computes
    if (s + 1 < nsteps) stageLoad(s + 1);
    if (s + 2 < nsteps) {           // prefetch HBM streams 2 steps ahead
      const bool sec2 = (FLAGS & F_DUAL) && (s + 2 >= steps1);
      const int kk2 = (sec2 ? (s + 2 - steps1) : (s + 2)) * BK;
      const void* A = sec2 ? A2 : A1;
      const unsigned short* Wt = sec2 ? Wt2 : Wt1;
      const size_t esz = (FLAGS & F_ABF16) ? 2 : 4;
      __builtin_prefetch((const char*)A + ((size_t)(rowBase + ar) * K + kk2) * esz, 0, 1);
      __builtin_prefetch(Wt + (size_t)(colBase + ar) * K + kk2, 0, 1);
    }

    // fragments + WMMA
    union Frag { uint4 u[2]; v16bf v; };
    const int fr  = lane & 15;
    const int kh  = (lane >> 4) * 8;    // A frag: K={kh..kh+7} and {16+kh..}
    const int kh2 = (lane >> 4) * 16;   // unused; B frag uses kh: K={kh..kh+7} and {16+kh..}
    Frag fa[2];
#pragma unroll
    for (int rt = 0; rt < 2; ++rt) {
      const int arow = wr * 32 + rt * 16 + fr;
      fa[rt].u[0] = *(const uint4*)&As[arow * BK + kh];
      fa[rt].u[1] = *(const uint4*)&As[arow * BK + 16 + kh];
    }
#pragma unroll
    for (int j = 0; j < 4; ++j) {
      Frag fb;
      const int bn = wc * 64 + j * 16 + fr;
      fb.u[0] = *(const uint4*)&Bs[bn * BK + kh];
      fb.u[1] = *(const uint4*)&Bs[bn * BK + 16 + kh];
#pragma unroll
      for (int rt = 0; rt < 2; ++rt)
        acc[rt][j] = __builtin_amdgcn_wmma_f32_16x16x32_bf16(
            false, fa[rt].v, false, fb.v, (short)0, acc[rt][j], false, false);
    }
    __syncthreads();
  }

  // ---- epilogue ----
#pragma unroll
  for (int rt = 0; rt < 2; ++rt) {
    const int rb = rowBase + wr * 32 + rt * 16 + ((lane >> 4) << 3);
#pragma unroll
    for (int j = 0; j < 4; ++j) {
      const int col = colBase + wc * 64 + j * 16 + (lane & 15);
      float bsum = 0.0f;
      if (bias1) bsum += bias1[col];
      if (bias2) bsum += bias2[col];
      const int oc = (FLAGS & F_Q2) ? ((col >> 5) * 64 + colsel * 32 + (col & 31)) : col;
#pragma unroll
      for (int e = 0; e < 8; ++e) {
        const int r = rb + e;
        float v = acc[rt][j][e] + bsum;
        if (FLAGS & F_RELU) v = fmaxf(v, 0.0f);
        if (FLAGS & F_MASK) v *= (rowmask[r] ? 1.0f : 0.0f);
        if (FLAGS & F_ADDPREV)
          v += bf2f(((const unsigned short*)outp)[(size_t)r * ldc + oc + 32]);
        if (FLAGS & F_OBF16)
          ((unsigned short*)outp)[(size_t)r * ldc + oc] = f2bf(v);
        else
          ((float*)outp)[(size_t)r * ldc + oc] = v;
      }
    }
  }
}

// ---------------------------------------------------------------------------
// Self-attention per (b,h): 64 queries x 64 keys x 32 dims (rows are n*B+b)
// ---------------------------------------------------------------------------
__global__ __launch_bounds__(64) void sa_attn_kernel(
    const float* __restrict__ q, const float* __restrict__ k,
    const float* __restrict__ v, float* __restrict__ out)
{
  const int b = blockIdx.x >> 3;
  const int h = blockIdx.x & 7;
  __shared__ float Kt[64 * 32];
  __shared__ float Vt[64 * 32];
  __shared__ float Sr[64 * 64];
  const int t = threadIdx.x;
  const size_t rowOff = ((size_t)(t * B_ + b)) * D_ + h * HD_;
  {
    const float4* kp = (const float4*)(k + rowOff);
    const float4* vp = (const float4*)(v + rowOff);
#pragma unroll
    for (int j = 0; j < 8; ++j) {
      *(float4*)&Kt[t * 32 + j * 4] = kp[j];
      *(float4*)&Vt[t * 32 + j * 4] = vp[j];
    }
  }
  float qr[32];
  {
    const float* qp = q + rowOff;
#pragma unroll
    for (int j = 0; j < 32; ++j) qr[j] = qp[j];
  }
  __syncthreads();

  float mx = -1e30f;
#pragma unroll 4
  for (int kk = 0; kk < 64; ++kk) {
    float d = 0.0f;
#pragma unroll
    for (int j = 0; j < 32; ++j) d += qr[j] * Kt[kk * 32 + j];
    d *= 0.17677669529663687f;            // 32^-0.5
    Sr[t * 64 + kk] = d;
    mx = fmaxf(mx, d);
  }
  float sum = 0.0f;
#pragma unroll 4
  for (int kk = 0; kk < 64; ++kk) {
    const float e = __expf(Sr[t * 64 + kk] - mx);
    Sr[t * 64 + kk] = e;
    sum += e;
  }
  const float inv = 1.0f / sum;
  float accv[32];
#pragma unroll
  for (int j = 0; j < 32; ++j) accv[j] = 0.0f;
#pragma unroll 4
  for (int kk = 0; kk < 64; ++kk) {
    const float a = Sr[t * 64 + kk] * inv;
#pragma unroll
    for (int j = 0; j < 32; ++j) accv[j] += a * Vt[kk * 32 + j];
  }
  float* op = out + rowOff;
#pragma unroll
  for (int j = 0; j < 32; ++j) op[j] = accv[j];
}

// ---------------------------------------------------------------------------
// Indexed local cross-attention: one wave per (query, head). wave32.
//   lanes own keys {lane, lane+32}; softmax via shfl; output dim = lane.
// ---------------------------------------------------------------------------
__global__ __launch_bounds__(256) void ca_attn_kernel(
    const unsigned short* __restrict__ q2, const unsigned short* __restrict__ k2,
    const unsigned short* __restrict__ vh, const int* __restrict__ index_pair,
    const int* __restrict__ ipb, const int* __restrict__ keyoff,
    float* __restrict__ out)
{
  const int lane = threadIdx.x & 31;
  const int wv   = threadIdx.x >> 5;
  const int pair = blockIdx.x * 8 + wv;
  const int qi = pair >> 3;
  const int h  = pair & 7;
  const int b  = ipb[qi];
  const int ko = keyoff[b];
  const int i0 = index_pair[qi * L_ + lane];
  const int i1 = index_pair[qi * L_ + 32 + lane];
  const int g0 = ko + (i0 > 0 ? i0 : 0);
  const int g1 = ko + (i1 > 0 ? i1 : 0);

  uint4 qv[8];
  {
    const uint4* qrow = (const uint4*)(q2 + (size_t)qi * 512 + h * 64);
#pragma unroll
    for (int i = 0; i < 8; ++i) qv[i] = qrow[i];
  }
  float s0 = -1e9f, s1 = -1e9f;
  if (i0 >= 0) {
    const uint4* kr = (const uint4*)(k2 + (size_t)g0 * 512 + h * 64);
    float d = 0.0f;
#pragma unroll
    for (int i = 0; i < 8; ++i) {
      const uint4 kvv = kr[i];
      d += bfdot2(qv[i].x, kvv.x) + bfdot2(qv[i].y, kvv.y)
         + bfdot2(qv[i].z, kvv.z) + bfdot2(qv[i].w, kvv.w);
    }
    s0 = d * 0.125f;                       // 64^-0.5
  }
  if (i1 >= 0) {
    const uint4* kr = (const uint4*)(k2 + (size_t)g1 * 512 + h * 64);
    float d = 0.0f;
#pragma unroll
    for (int i = 0; i < 8; ++i) {
      const uint4 kvv = kr[i];
      d += bfdot2(qv[i].x, kvv.x) + bfdot2(qv[i].y, kvv.y)
         + bfdot2(qv[i].z, kvv.z) + bfdot2(qv[i].w, kvv.w);
    }
    s1 = d * 0.125f;
  }
  float m = fmaxf(s0, s1);
#pragma unroll
  for (int o = 16; o > 0; o >>= 1) m = fmaxf(m, __shfl_xor(m, o, 32));
  float p0 = __expf(s0 - m);
  float p1 = __expf(s1 - m);
  float sm = p0 + p1;
#pragma unroll
  for (int o = 16; o > 0; o >>= 1) sm += __shfl_xor(sm, o, 32);
  const float inv = 1.0f / sm;
  p0 *= inv; p1 *= inv;

  float accd = 0.0f;
  const unsigned short* vbase = vh + h * HD_ + lane;
#pragma unroll 4
  for (int kk = 0; kk < 32; ++kk) {
    int gk = __shfl(g0, kk, 32);
    float a = __shfl(p0, kk, 32);
    accd += a * bf2f(vbase[(size_t)gk * D_]);
    gk = __shfl(g1, kk, 32);
    a = __shfl(p1, kk, 32);
    accd += a * bf2f(vbase[(size_t)gk * D_]);
  }
  out[(size_t)qi * D_ + h * HD_ + lane] = accd;
}

// ---------------------------------------------------------------------------
// out[r] = LayerNorm(x[pr] + resid[pr]) * gw + gb     (wave per row, D=256)
// ---------------------------------------------------------------------------
__global__ __launch_bounds__(256) void ln_res_kernel(
    const float* __restrict__ x, const float* __restrict__ resid,
    const float* __restrict__ gw, const float* __restrict__ gb,
    float* __restrict__ out, int permIn)
{
  const int lane = threadIdx.x & 31;
  const int wv = threadIdx.x >> 5;
  const int r = blockIdx.x * 8 + wv;
  const int pr = permIn ? ((r & 63) * 64 + (r >> 6)) : r;
  const float* xp = x + (size_t)pr * D_ + lane * 8;
  const float* rp = resid + (size_t)pr * D_ + lane * 8;
  float vals[8];
  {
    const float4 a0 = *(const float4*)xp;
    const float4 a1 = *(const float4*)(xp + 4);
    const float4 b0 = *(const float4*)rp;
    const float4 b1 = *(const float4*)(rp + 4);
    vals[0] = a0.x + b0.x; vals[1] = a0.y + b0.y;
    vals[2] = a0.z + b0.z; vals[3] = a0.w + b0.w;
    vals[4] = a1.x + b1.x; vals[5] = a1.y + b1.y;
    vals[6] = a1.z + b1.z; vals[7] = a1.w + b1.w;
  }
  float s = 0.0f;
#pragma unroll
  for (int i = 0; i < 8; ++i) s += vals[i];
#pragma unroll
  for (int o = 16; o > 0; o >>= 1) s += __shfl_xor(s, o, 32);
  const float mean = s * (1.0f / 256.0f);
  float vs = 0.0f;
#pragma unroll
  for (int i = 0; i < 8; ++i) { const float d = vals[i] - mean; vs += d * d; }
#pragma unroll
  for (int o = 16; o > 0; o >>= 1) vs += __shfl_xor(vs, o, 32);
  const float rs = rsqrtf(vs * (1.0f / 256.0f) + 1e-5f);
  float* op = out + (size_t)r * D_ + lane * 8;
  const float* wp = gw + lane * 8;
  const float* bp = gb + lane * 8;
#pragma unroll
  for (int i = 0; i < 8; ++i) op[i] = (vals[i] - mean) * rs * wp[i] + bp[i];
}

// ---------------------------------------------------------------------------
// Host launch
// ---------------------------------------------------------------------------
extern "C" void kernel_launch(void* const* d_in, const int* in_sizes, int n_in,
                              void* d_out, int out_size, void* d_ws, size_t ws_size,
                              hipStream_t stream) {
  (void)in_sizes; (void)n_in; (void)out_size; (void)ws_size;

  const float* tgt        = (const float*)d_in[0];
  const float* memory     = (const float*)d_in[1];
  const float* pos        = (const float*)d_in[2];
  const float* query_pos  = (const float*)d_in[3];
  const float* qsine      = (const float*)d_in[4];
  const int*   key_cnt    = (const int*)d_in[5];
  const int*   index_pair = (const int*)d_in[6];
  const int*   ipb        = (const int*)d_in[7];
  const unsigned char* vmask = (const unsigned char*)d_in[8];   // jnp bool = 1 byte

  auto Wf = [&](int i) { return (const float*)d_in[i]; };

  // --- workspace carve ---
  char* ws = (char*)d_ws;
  size_t off = 0;
  auto alloc = [&](size_t bytes) -> char* {
    char* p = ws + off;
    off = (off + bytes + 255) & ~(size_t)255;
    return p;
  };

  // bf16 transposed weights: 13 x D*D, linear1 (K=256,N=1024), linear2 (K=1024,N=256)
  const int wsrc[15] = {9, 11, 13, 15, 17, 19, 21, 23, 25, 27, 29, 31, 33, 35, 37};
  const int wK[15]   = {256,256,256,256,256,256,256,256,256,256,256,256,256, 256,1024};
  const int wN[15]   = {256,256,256,256,256,256,256,256,256,256,256,256,256,1024, 256};
  unsigned short* Wt[15];
  for (int i = 0; i < 15; ++i)
    Wt[i] = (unsigned short*)alloc((size_t)wK[i] * wN[i] * sizeof(unsigned short));

  float* q_sa    = (float*)alloc((size_t)TQ_ * D_ * 4);
  float* k_sa    = (float*)alloc((size_t)TQ_ * D_ * 4);
  float* v_sa    = (float*)alloc((size_t)TQ_ * D_ * 4);
  float* sa_ctx  = (float*)alloc((size_t)TQ_ * D_ * 4);
  float* sa_proj = (float*)alloc((size_t)TQ_ * D_ * 4);
  float* tgt2    = (float*)alloc((size_t)TQ_ * D_ * 4);   // post-norm1, bn order
  float* ca_ctx  = (float*)alloc((size_t)TQ_ * D_ * 4);
  float* ca_proj = (float*)alloc((size_t)TQ_ * D_ * 4);
  float* tgt3    = (float*)alloc((size_t)TQ_ * D_ * 4);   // post-norm2
  float* ffn_out = (float*)alloc((size_t)TQ_ * D_ * 4);
  unsigned short* q2     = (unsigned short*)alloc((size_t)TQ_ * 512 * 2);
  unsigned short* k2     = (unsigned short*)alloc((size_t)TK_ * 512 * 2);
  unsigned short* vh     = (unsigned short*)alloc((size_t)TK_ * D_ * 2);
  unsigned short* hidden = (unsigned short*)alloc((size_t)TQ_ * FF_ * 2);
  int* key_off = (int*)alloc(B_ * sizeof(int));

  // --- weight prep + prefix sums ---
  for (int i = 0; i < 15; ++i) {
    const int n = wK[i] * wN[i];
    wconv_kernel<<<(n + 255) / 256, 256, 0, stream>>>(Wf(wsrc[i]), Wt[i], wK[i], wN[i]);
  }
  prefix_kernel<<<1, B_, 0, stream>>>(key_cnt, key_off);

  const dim3 g2(2, TQ_ / BM);    // N=256,  M=4096
  const dim3 g2k(2, TK_ / BM);   // N=256,  M=49152
  const dim3 g8(8, TQ_ / BM);    // N=1024, M=4096

  // --- self-attention projections (rows in nb order) ---
  gemm_wmma_bf16<F_DUAL><<<g2, 256, 0, stream>>>(tgt, query_pos, Wt[0], Wt[1],
      Wf(10), Wf(12), nullptr, q_sa, D_, 0, D_);
  gemm_wmma_bf16<F_DUAL><<<g2, 256, 0, stream>>>(tgt, query_pos, Wt[2], Wt[3],
      Wf(14), Wf(16), nullptr, k_sa, D_, 0, D_);
  gemm_wmma_bf16<0><<<g2, 256, 0, stream>>>(tgt, nullptr, Wt[4], nullptr,
      Wf(18), nullptr, nullptr, v_sa, D_, 0, D_);
  sa_attn_kernel<<<B_ * H_, 64, 0, stream>>>(q_sa, k_sa, v_sa, sa_ctx);
  gemm_wmma_bf16<0><<<g2, 256, 0, stream>>>(sa_ctx, nullptr, Wt[5], nullptr,
      Wf(20), nullptr, nullptr, sa_proj, D_, 0, D_);
  // norm1: tgt + sa_proj (nb order in) -> tgt2 (bn order out)
  ln_res_kernel<<<TQ_ / 8, 256, 0, stream>>>(sa_proj, tgt, Wf(39), Wf(40), tgt2, 1);

  // --- cross-attention projections ---
  // q2 lo-half: tgt2@Wqc + query_pos(perm)@Wqp ; q2 hi-half: qsine(perm)@Wqps
  gemm_wmma_bf16<F_DUAL | F_PERMA2 | F_Q2 | F_OBF16><<<g2, 256, 0, stream>>>(
      tgt2, query_pos, Wt[6], Wt[7], Wf(22), Wf(24), nullptr, q2, D_, 0, 512);
  gemm_wmma_bf16<F_PERMA1 | F_Q2 | F_OBF16><<<g2, 256, 0, stream>>>(
      qsine, nullptr, Wt[11], nullptr, Wf(32), nullptr, nullptr, q2, D_, 1, 512);
  // k2 hi-half: masked k_pos ; k2 lo-half: masked k_content + (stored k_pos half)
  gemm_wmma_bf16<F_Q2 | F_OBF16 | F_MASK><<<g2k, 256, 0, stream>>>(
      pos, nullptr, Wt[9], nullptr, Wf(28), nullptr, vmask, k2, D_, 1, 512);
  gemm_wmma_bf16<F_Q2 | F_OBF16 | F_MASK | F_ADDPREV><<<g2k, 256, 0, stream>>>(
      memory, nullptr, Wt[8], nullptr, Wf(26), nullptr, vmask, k2, D_, 0, 512);
  gemm_wmma_bf16<F_OBF16 | F_MASK><<<g2k, 256, 0, stream>>>(
      memory, nullptr, Wt[10], nullptr, Wf(30), nullptr, vmask, vh, D_, 0, D_);

  ca_attn_kernel<<<(TQ_ * H_) / 8, 256, 0, stream>>>(q2, k2, vh, index_pair, ipb,
                                                     key_off, ca_ctx);
  gemm_wmma_bf16<0><<<g2, 256, 0, stream>>>(ca_ctx, nullptr, Wt[12], nullptr,
      Wf(34), nullptr, nullptr, ca_proj, D_, 0, D_);
  ln_res_kernel<<<TQ_ / 8, 256, 0, stream>>>(ca_proj, tgt2, Wf(41), Wf(42), tgt3, 0);

  // --- FFN ---
  gemm_wmma_bf16<F_RELU | F_OBF16><<<g8, 256, 0, stream>>>(tgt3, nullptr, Wt[13],
      nullptr, Wf(36), nullptr, nullptr, hidden, D_, 0, FF_);
  gemm_wmma_bf16<F_ABF16><<<g2, 256, 0, stream>>>(hidden, nullptr, Wt[14], nullptr,
      Wf(38), nullptr, nullptr, ffn_out, FF_, 0, D_);
  ln_res_kernel<<<TQ_ / 8, 256, 0, stream>>>(ffn_out, tgt3, Wf(43), Wf(44),
                                             (float*)d_out, 0);
}
